// Attention_EncDec_27075473834737
// MI455X (gfx1250) — matrix operand order repaired and checked by reading, not compiled, not run on hardware
//
#include <hip/hip_runtime.h>

typedef __attribute__((ext_vector_type(16))) _Float16 v16h;
typedef __attribute__((ext_vector_type(8)))  _Float16 v8h;
typedef __attribute__((ext_vector_type(8)))  float    v8f;

#define S_LEN   4096
#define IN_DIM  512
#define H_DIM   2048
#define OUT_DIM 512
#define G3      (3 * H_DIM)   // 6144
#define X_DIM   (2 * H_DIM)   // 4096

#define DEC_BLOCKS 32   // 32*256 = 8192 threads

__device__ __forceinline__ float sigmoidf_(float x) { return 1.0f / (1.0f + expf(-x)); }

// ---------------- software grid barrier (sense-reversing) ----------------
__device__ __forceinline__ void grid_barrier(unsigned* bar) {
  __syncthreads();
  if (threadIdx.x == 0) {
    __threadfence();
    unsigned nb = gridDim.x;
    unsigned g = __hip_atomic_load(&bar[1], __ATOMIC_ACQUIRE, __HIP_MEMORY_SCOPE_AGENT);
    unsigned arrived =
        __hip_atomic_fetch_add(&bar[0], 1u, __ATOMIC_ACQ_REL, __HIP_MEMORY_SCOPE_AGENT);
    if (arrived == nb - 1) {
      __hip_atomic_store(&bar[0], 0u, __ATOMIC_RELAXED, __HIP_MEMORY_SCOPE_AGENT);
      __hip_atomic_fetch_add(&bar[1], 1u, __ATOMIC_RELEASE, __HIP_MEMORY_SCOPE_AGENT);
    } else {
      while (__hip_atomic_load(&bar[1], __ATOMIC_ACQUIRE, __HIP_MEMORY_SCOPE_AGENT) == g)
        __builtin_amdgcn_s_sleep(2);
    }
    __threadfence();
  }
  __syncthreads();
}

// ---------------- init: barrier counters + h0 (f32 and f16 shadow) ----------------
__global__ void init_ws_kernel(unsigned* __restrict__ bar, float* __restrict__ h,
                               _Float16* __restrict__ h16) {
  int t = threadIdx.x;
  if (t < 64) bar[t] = 0u;
  for (int i = t; i < H_DIM; i += blockDim.x) { h[i] = 0.0f; h16[i] = (_Float16)0.0f; }
}

// ---------------- fp32 -> f16 conversion (grid-stride) ----------------
__global__ void f32_to_f16_kernel(const float* __restrict__ src, _Float16* __restrict__ dst,
                                  size_t n) {
  size_t i = (size_t)blockIdx.x * blockDim.x + threadIdx.x;
  size_t stride = (size_t)gridDim.x * blockDim.x;
  for (; i < n; i += stride) dst[i] = (_Float16)src[i];
}

// ---------------- WMMA GEMM: C[M,N] = A[M,K] * W[N,K]^T + bias[N] ----------------
// Register-blocked 32x32 strip per wave = 2 M-tiles x 2 N-tiles (4 wmma / k32).
// 4 accumulators (32 VGPRs) + fragments stays spill-free even with 2x unroll/pipelining.
__global__ void __launch_bounds__(256)
wmma_gemm_bias_kernel(const _Float16* __restrict__ A,  // M x K row-major
                      const _Float16* __restrict__ W,  // N x K row-major
                      const float* __restrict__ bias,  // N
                      float* __restrict__ C,           // M x N row-major
                      int M, int N, int K) {
  int gt   = blockIdx.x * blockDim.x + threadIdx.x;
  int wave = gt >> 5;
  int lane = threadIdx.x & 31;
  int stripsN = N >> 5;
  int sm = wave / stripsN;
  int sn = wave - sm * stripsN;
  if (sm * 32 >= M) return;
  int m0 = sm << 5, n0 = sn << 5;
  int lmod = lane & 15;
  int hi   = lane >> 4;

  const _Float16* arow0 = A + (size_t)(m0 + lmod) * K;
  const _Float16* arow1 = A + (size_t)(m0 + 16 + lmod) * K;
  const _Float16* wrow0 = W + (size_t)(n0 + lmod) * K;
  const _Float16* wrow1 = W + (size_t)(n0 + 16 + lmod) * K;

  v8f zero = {};
  v8f acc00 = zero, acc01 = zero, acc10 = zero, acc11 = zero;

  for (int k0 = 0; k0 < K; k0 += 32) {
    v16h a0, a1;
    {
      v8h p00 = *(const v8h*)(arow0 + k0 + hi * 16);
      v8h p01 = *(const v8h*)(arow0 + k0 + 8 + hi * 16);
      v8h p10 = *(const v8h*)(arow1 + k0 + hi * 16);
      v8h p11 = *(const v8h*)(arow1 + k0 + 8 + hi * 16);
#pragma unroll
      for (int e = 0; e < 8; ++e) {
        a0[e] = p00[e]; a0[e + 8] = p01[e];
        a1[e] = p10[e]; a1[e + 8] = p11[e];
      }
    }
    v16h b0 = *(const v16h*)(wrow0 + k0 + hi * 16);
    v16h b1 = *(const v16h*)(wrow1 + k0 + hi * 16);
    acc00 = __builtin_amdgcn_wmma_f32_16x16x32_f16(false, a0, false, b0, (short)0, acc00,
                                                   false, false);
    acc01 = __builtin_amdgcn_wmma_f32_16x16x32_f16(false, a0, false, b1, (short)0, acc01,
                                                   false, false);
    acc10 = __builtin_amdgcn_wmma_f32_16x16x32_f16(false, a1, false, b0, (short)0, acc10,
                                                   false, false);
    acc11 = __builtin_amdgcn_wmma_f32_16x16x32_f16(false, a1, false, b1, (short)0, acc11,
                                                   false, false);
  }

  int ncol0 = n0 + lmod;
  int ncol1 = n0 + 16 + lmod;
  float bv0 = bias[ncol0];
  float bv1 = bias[ncol1];
#pragma unroll
  for (int v = 0; v < 8; ++v) {
    int mA = m0 + v + 8 * hi;        // M tile 0 rows
    int mB = m0 + 16 + v + 8 * hi;   // M tile 1 rows
    C[(size_t)mA * N + ncol0] = acc00[v] + bv0;
    C[(size_t)mA * N + ncol1] = acc01[v] + bv1;
    C[(size_t)mB * N + ncol0] = acc10[v] + bv0;
    C[(size_t)mB * N + ncol1] = acc11[v] + bv1;
  }
}

// ---------------- persistent encoder GRU: recurrence on the matrix cores ----------------
// 128 waves; wave w owns h[16w..16w+16) and its 3 gate rows. h (f16 shadow) is broadcast
// into all 16 B columns, so D's every column = Whh_tile . h -> no cross-wave gh exchange,
// one grid barrier per timestep.
__global__ void encoder_gru_kernel(const float* __restrict__ Gi,      // S x 3H (precomputed)
                                   const _Float16* __restrict__ Whh,  // 3H x H (f16, L2-resident)
                                   const float* __restrict__ bhh,     // 3H
                                   float* __restrict__ h,             // H (f32 master)
                                   _Float16* __restrict__ h16,        // H (f16 shadow)
                                   float* __restrict__ enc_out,       // S x H (f32)
                                   _Float16* __restrict__ enc_out_h,  // S x H (f16)
                                   unsigned* __restrict__ bar) {
  int gt   = blockIdx.x * blockDim.x + threadIdx.x;
  int wave = gt >> 5;          // 0..127
  int lane = threadIdx.x & 31;
  int lmod = lane & 15;
  int hi   = lane >> 4;
  int i0   = wave << 4;        // first h element of this wave's slice

  // A-fragment row pointers for the three gate tiles (per-lane row = base + lmod)
  const _Float16* wr_r = Whh + (size_t)(i0 + lmod) * H_DIM;
  const _Float16* wr_z = Whh + (size_t)(i0 + H_DIM + lmod) * H_DIM;
  const _Float16* wr_n = Whh + (size_t)(i0 + 2 * H_DIM + lmod) * H_DIM;

  v8f zero = {};
  for (int t = 0; t < S_LEN; ++t) {
    v8f ar = zero, az = zero, an = zero;
#pragma unroll 4
    for (int k0 = 0; k0 < H_DIM; k0 += 32) {
      // broadcast B: all 16 columns carry the same h[k0+16hi .. +16) values
      v16h b = *(const v16h*)(h16 + k0 + hi * 16);
      v16h fr, fz, fn;
      {
        v8h r0 = *(const v8h*)(wr_r + k0 + hi * 16);
        v8h r1 = *(const v8h*)(wr_r + k0 + 8 + hi * 16);
        v8h z0 = *(const v8h*)(wr_z + k0 + hi * 16);
        v8h z1 = *(const v8h*)(wr_z + k0 + 8 + hi * 16);
        v8h n0 = *(const v8h*)(wr_n + k0 + hi * 16);
        v8h n1 = *(const v8h*)(wr_n + k0 + 8 + hi * 16);
#pragma unroll
        for (int e = 0; e < 8; ++e) {
          fr[e] = r0[e]; fr[e + 8] = r1[e];
          fz[e] = z0[e]; fz[e + 8] = z1[e];
          fn[e] = n0[e]; fn[e + 8] = n1[e];
        }
      }
      ar = __builtin_amdgcn_wmma_f32_16x16x32_f16(false, fr, false, b, (short)0, ar, false, false);
      az = __builtin_amdgcn_wmma_f32_16x16x32_f16(false, fz, false, b, (short)0, az, false, false);
      an = __builtin_amdgcn_wmma_f32_16x16x32_f16(false, fn, false, b, (short)0, an, false, false);
    }
    // D layout: lane (n=lmod, hi), VGPR v -> row v + 8*hi; all 16 columns identical.
    // Lanes with lmod==0 update 8 h-elements each.
    if (lmod == 0) {
      const float* gi = Gi + (size_t)t * G3;
#pragma unroll
      for (int v = 0; v < 8; ++v) {
        int i = i0 + hi * 8 + v;
        float ghr = ar[v] + bhh[i];
        float ghz = az[v] + bhh[i + H_DIM];
        float ghn = an[v] + bhh[i + 2 * H_DIM];
        float r  = sigmoidf_(gi[i] + ghr);
        float z  = sigmoidf_(gi[i + H_DIM] + ghz);
        float n  = tanhf(gi[i + 2 * H_DIM] + r * ghn);
        float hn = (1.0f - z) * n + z * h[i];
        h[i]   = hn;
        h16[i] = (_Float16)hn;
        enc_out[(size_t)t * H_DIM + i]   = hn;
        enc_out_h[(size_t)t * H_DIM + i] = (_Float16)hn;
      }
    }
    grid_barrier(bar);  // h/h16 published for everyone's next-step B fragments
  }
}

// ---------------- persistent attention decoder ----------------
__global__ void decoder_kernel(const float* __restrict__ U,        // S x H
                               const float* __restrict__ enc_out,  // S x H
                               float* __restrict__ h,              // H (from encoder)
                               const float* __restrict__ W_w, const float* __restrict__ W_b,
                               const float* __restrict__ att_w, const float* __restrict__ att_b,
                               const _Float16* __restrict__ dWih16,  // 3H x 2H (f16)
                               const float* __restrict__ dec_bih,
                               const _Float16* __restrict__ dWhh16,  // 3H x H (f16)
                               const float* __restrict__ dec_bhh,
                               const float* __restrict__ o2h_w,    // H x OUT
                               const float* __restrict__ o2h_b,
                               const float* __restrict__ h2o_w,    // OUT x H
                               const float* __restrict__ h2o_b,
                               const int* __restrict__ Tptr, float* __restrict__ out,
                               float* __restrict__ Wh, float* __restrict__ scores,
                               float* __restrict__ x, float* __restrict__ gi2,
                               float* __restrict__ gh2, float* __restrict__ logits,
                               float* __restrict__ red, unsigned* __restrict__ bar) {
  __shared__ float smem[H_DIM + X_DIM];  // 24 KB operand cache
  __shared__ float sv[256];
  __shared__ int   si[256];
  int tid  = blockIdx.x * blockDim.x + threadIdx.x;
  int ltid = threadIdx.x;
  int T = Tptr[0];
  int* ridx = (int*)(red + 16);

  for (int step = 0; step < T; ++step) {
    // --- A: Wh = W_w . h + W_b ---
    for (int i = ltid; i < H_DIM; i += blockDim.x) smem[i] = h[i];
    __syncthreads();
    if (tid < H_DIM) {
      const float* wr = W_w + (size_t)tid * H_DIM;
      float a = W_b[tid];
      for (int k = 0; k < H_DIM; ++k) a = fmaf(wr[k], smem[k], a);
      Wh[tid] = a;
    }
    grid_barrier(bar);

    // --- B: scores[s] = tanh(U[s,:] + Wh) . att_w + att_b ---
    for (int i = ltid; i < H_DIM; i += blockDim.x) {
      smem[i]         = Wh[i];
      smem[H_DIM + i] = att_w[i];
    }
    __syncthreads();
    if (tid < S_LEN) {
      const float* Ur = U + (size_t)tid * H_DIM;
      float a = att_b[0];
      for (int k = 0; k < H_DIM; ++k) a = fmaf(tanhf(Ur[k] + smem[k]), smem[H_DIM + k], a);
      scores[tid] = a;
    }
    grid_barrier(bar);

    // --- C1: block0 computes softmax max + sum ---
    if (blockIdx.x == 0) {
      float m = -3.4e38f;
      for (int s = ltid; s < S_LEN; s += blockDim.x) m = fmaxf(m, scores[s]);
      sv[ltid] = m; __syncthreads();
      for (int st = 128; st > 0; st >>= 1) {
        if (ltid < st) sv[ltid] = fmaxf(sv[ltid], sv[ltid + st]);
        __syncthreads();
      }
      float mx = sv[0]; __syncthreads();
      float ssum = 0.0f;
      for (int s = ltid; s < S_LEN; s += blockDim.x) ssum += expf(scores[s] - mx);
      sv[ltid] = ssum; __syncthreads();
      for (int st = 128; st > 0; st >>= 1) {
        if (ltid < st) sv[ltid] += sv[ltid + st];
        __syncthreads();
      }
      if (ltid == 0) { red[2] = mx; red[3] = 1.0f / sv[0]; }
    }
    grid_barrier(bar);

    // --- C2: attw (in place) ---
    if (tid < S_LEN) scores[tid] = expf(scores[tid] - red[2]) * red[3];
    grid_barrier(bar);

    // --- D+E: context (coalesced column sweep) and x build ---
    if (tid < H_DIM) {
      float c = 0.0f;
      for (int s = 0; s < S_LEN; ++s) c = fmaf(scores[s], enc_out[(size_t)s * H_DIM + tid], c);
      x[H_DIM + tid] = c;
      float xf = o2h_b[tid];
      if (step > 0) xf += o2h_w[(size_t)tid * OUT_DIM + ridx[0]];  // one-hot = column gather
      x[tid] = xf;
    }
    grid_barrier(bar);

    // --- F: gi2 = dec_Wih . x + bih ; gh2 = dec_Whh . h + bhh (f16 weight streams) ---
    for (int i = ltid; i < H_DIM; i += blockDim.x) smem[i] = h[i];
    for (int i = ltid; i < X_DIM; i += blockDim.x) smem[H_DIM + i] = x[i];
    __syncthreads();
    if (tid < G3) {
      const _Float16* wi = dWih16 + (size_t)tid * X_DIM;
      float a = dec_bih[tid];
      for (int k = 0; k < X_DIM; k += 8) {
        v8h wv = *(const v8h*)(wi + k);
#pragma unroll
        for (int e = 0; e < 8; ++e) a = fmaf((float)wv[e], smem[H_DIM + k + e], a);
      }
      gi2[tid] = a;
      const _Float16* wh = dWhh16 + (size_t)tid * H_DIM;
      float b = dec_bhh[tid];
      for (int k = 0; k < H_DIM; k += 8) {
        v8h wv = *(const v8h*)(wh + k);
#pragma unroll
        for (int e = 0; e < 8; ++e) b = fmaf((float)wv[e], smem[k + e], b);
      }
      gh2[tid] = b;
    }
    grid_barrier(bar);

    // --- G: GRU gate update ---
    if (tid < H_DIM) {
      float r = sigmoidf_(gi2[tid] + gh2[tid]);
      float z = sigmoidf_(gi2[tid + H_DIM] + gh2[tid + H_DIM]);
      float n = tanhf(gi2[tid + 2 * H_DIM] + r * gh2[tid + 2 * H_DIM]);
      h[tid] = (1.0f - z) * n + z * h[tid];
    }
    grid_barrier(bar);

    // --- H: logits = h2o_w . h + h2o_b ---
    for (int i = ltid; i < H_DIM; i += blockDim.x) smem[i] = h[i];
    __syncthreads();
    if (tid < OUT_DIM) {
      const float* wr = h2o_w + (size_t)tid * H_DIM;
      float a = h2o_b[tid];
      for (int k = 0; k < H_DIM; ++k) a = fmaf(wr[k], smem[k], a);
      logits[tid] = a;
    }
    grid_barrier(bar);

    // --- I: block0 argmax + logsumexp ---
    if (blockIdx.x == 0) {
      float m = -3.4e38f; int bi = OUT_DIM;
      for (int o = ltid; o < OUT_DIM; o += blockDim.x) {
        float v = logits[o];
        if (v > m) { m = v; bi = o; }
      }
      sv[ltid] = m; si[ltid] = bi; __syncthreads();
      for (int st = 128; st > 0; st >>= 1) {
        if (ltid < st) {
          float ov = sv[ltid + st]; int oi = si[ltid + st];
          if (ov > sv[ltid] || (ov == sv[ltid] && oi < si[ltid])) { sv[ltid] = ov; si[ltid] = oi; }
        }
        __syncthreads();
      }
      float mx = sv[0]; int am = si[0]; __syncthreads();
      float ssum = 0.0f;
      for (int o = ltid; o < OUT_DIM; o += blockDim.x) ssum += expf(logits[o] - mx);
      sv[ltid] = ssum; __syncthreads();
      for (int st = 128; st > 0; st >>= 1) {
        if (ltid < st) sv[ltid] += sv[ltid + st];
        __syncthreads();
      }
      if (ltid == 0) { red[0] = mx; red[1] = logf(sv[0]); ridx[0] = am; }
    }
    grid_barrier(bar);

    // --- J: log-softmax output ---
    if (tid < OUT_DIM) out[(size_t)step * OUT_DIM + tid] = logits[tid] - red[0] - red[1];
    grid_barrier(bar);
  }
}

// ---------------- host launch ----------------
extern "C" void kernel_launch(void* const* d_in, const int* in_sizes, int n_in,
                              void* d_out, int out_size, void* d_ws, size_t ws_size,
                              hipStream_t stream) {
  (void)in_sizes; (void)n_in; (void)out_size; (void)ws_size;
  const float* in_value = (const float*)d_in[0];
  const float* enc_Wih  = (const float*)d_in[1];
  const float* enc_Whh  = (const float*)d_in[2];
  const float* enc_bih  = (const float*)d_in[3];
  const float* enc_bhh  = (const float*)d_in[4];
  const float* dec_Wih  = (const float*)d_in[5];
  const float* dec_Whh  = (const float*)d_in[6];
  const float* dec_bih  = (const float*)d_in[7];
  const float* dec_bhh  = (const float*)d_in[8];
  const float* U_w      = (const float*)d_in[9];
  const float* U_b      = (const float*)d_in[10];
  const float* att_w    = (const float*)d_in[11];
  const float* att_b    = (const float*)d_in[12];
  const float* W_w      = (const float*)d_in[13];
  const float* W_b      = (const float*)d_in[14];
  const float* o2h_w    = (const float*)d_in[15];
  const float* o2h_b    = (const float*)d_in[16];
  const float* h2o_w    = (const float*)d_in[17];
  const float* h2o_b    = (const float*)d_in[18];
  const int*   Tptr     = (const int*)d_in[19];
  float* out = (float*)d_out;

  char* ws = (char*)d_ws;
  size_t off = 0;
  auto take = [&](size_t bytes) -> char* {
    char* p = ws + off;
    off = (off + bytes + 255) & ~(size_t)255;
    return p;
  };

  unsigned* bar       = (unsigned*)take(256);
  float*    h         = (float*)take(H_DIM * 4);
  _Float16* h16       = (_Float16*)take(H_DIM * 2);
  float*    red       = (float*)take(64 * 4);
  float*    Wh        = (float*)take(H_DIM * 4);
  float*    scores    = (float*)take(S_LEN * 4);
  float*    xbuf      = (float*)take(X_DIM * 4);
  float*    gi2       = (float*)take(G3 * 4);
  float*    gh2       = (float*)take(G3 * 4);
  float*    logits    = (float*)take(OUT_DIM * 4);
  float*    Gi        = (float*)take((size_t)S_LEN * G3 * 4);        // 100.7 MB
  float*    enc_out   = (float*)take((size_t)S_LEN * H_DIM * 4);     // 33.6 MB
  _Float16* enc_out_h = (_Float16*)take((size_t)S_LEN * H_DIM * 2);  // 16.8 MB
  float*    Ubuf      = (float*)take((size_t)S_LEN * H_DIM * 4);     // 33.6 MB
  _Float16* in16      = (_Float16*)take((size_t)S_LEN * IN_DIM * 2);
  _Float16* Wih16     = (_Float16*)take((size_t)G3 * IN_DIM * 2);
  _Float16* Whh16     = (_Float16*)take((size_t)G3 * H_DIM * 2);     // 25 MB, L2-resident
  _Float16* Uw16      = (_Float16*)take((size_t)H_DIM * H_DIM * 2);
  _Float16* dWih16    = (_Float16*)take((size_t)G3 * X_DIM * 2);     // 50 MB
  _Float16* dWhh16    = (_Float16*)take((size_t)G3 * H_DIM * 2);     // 25 MB

  // init barrier + h0
  init_ws_kernel<<<1, 256, 0, stream>>>(bar, h, h16);

  // fp32 -> f16 operand conversions
  f32_to_f16_kernel<<<2048, 256, 0, stream>>>(in_value, in16, (size_t)S_LEN * IN_DIM);
  f32_to_f16_kernel<<<2048, 256, 0, stream>>>(enc_Wih, Wih16, (size_t)G3 * IN_DIM);
  f32_to_f16_kernel<<<4096, 256, 0, stream>>>(enc_Whh, Whh16, (size_t)G3 * H_DIM);
  f32_to_f16_kernel<<<2048, 256, 0, stream>>>(U_w, Uw16, (size_t)H_DIM * H_DIM);
  f32_to_f16_kernel<<<8192, 256, 0, stream>>>(dec_Wih, dWih16, (size_t)G3 * X_DIM);
  f32_to_f16_kernel<<<4096, 256, 0, stream>>>(dec_Whh, dWhh16, (size_t)G3 * H_DIM);

  // Gi = in_value @ enc_Wih^T + enc_bih   (M=4096, N=6144, K=512) — WMMA
  {
    int M = S_LEN, N = G3, K = IN_DIM;
    int strips = (M / 32) * (N / 32);       // 24576 waves
    wmma_gemm_bias_kernel<<<strips / 8, 256, 0, stream>>>(in16, Wih16, enc_bih, Gi, M, N, K);
  }

  // sequential encoder recurrence on matrix cores (persistent, 1 barrier/step)
  encoder_gru_kernel<<<64, 64, 0, stream>>>(Gi, Whh16, enc_bhh, h, h16, enc_out, enc_out_h,
                                            bar);

  // U = enc_out @ U_w^T + U_b   (M=4096, N=2048, K=2048) — WMMA
  {
    int M = S_LEN, N = H_DIM, K = H_DIM;
    int strips = (M / 32) * (N / 32);       // 8192 waves
    wmma_gemm_bias_kernel<<<strips / 8, 256, 0, stream>>>(enc_out_h, Uw16, U_b, Ubuf, M, N, K);
  }

  // greedy attention decoder (persistent, grid-synced), writes log-softmax rows to d_out
  decoder_kernel<<<DEC_BLOCKS, 256, 0, stream>>>(Ubuf, enc_out, h, W_w, W_b, att_w, att_b,
                                                 dWih16, dec_bih, dWhh16, dec_bhh, o2h_w,
                                                 o2h_b, h2o_w, h2o_b, Tptr, out, Wh, scores,
                                                 xbuf, gi2, gh2, logits, red, bar);
}
